// GTMGCBlock_18545668784204
// MI455X (gfx1250) — compile-verified, matrix operand order checked by reading, not run on hardware
//
#include <hip/hip_runtime.h>
#include <hip/hip_bf16.h>

// ---------------------------------------------------------------------------
// GTMGC transformer block for MI455X (gfx1250), wave32 + WMMA f16->f32,
// async global->LDS staging (ASYNCcnt) for shared operand tiles via inline
// CDNA5 asm (GLOBAL_LOAD_ASYNC_TO_LDS_B128 + S_WAIT_ASYNCCNT).
// B=256, N=128, D_MODEL=256, N_HEAD=8, HEAD_DIM=32, D_FFN=1024.
// Workspace requirement: ~237 MB.
// ---------------------------------------------------------------------------

typedef __attribute__((ext_vector_type(16))) _Float16 v16h;
typedef __attribute__((ext_vector_type(8)))  _Float16 v8h;
typedef __attribute__((ext_vector_type(8)))  float    v8f;

#define BZ 256
#define NN 128
#define DM 256
#define NH 8
#define HD 32
#define DF 1024

// Copy one 16-byte chunk global -> LDS through the async DMA path (ASYNCcnt).
// A generic pointer to LDS holds the wave-relative LDS byte offset in its low
// 32 bits (ISA 10.2), which is exactly the VDST
// operand of GLOBAL_LOAD_ASYNC_TO_LDS_*.
__device__ __forceinline__ void stage16(const _Float16* g, _Float16* l) {
  const unsigned lds_off = (unsigned)(unsigned long long)l;
  asm volatile("global_load_async_to_lds_b128 %0, %1, off"
               :
               : "v"(lds_off), "v"((const void*)g)
               : "memory");
}

__device__ __forceinline__ void async_wait0() {
#if __has_builtin(__builtin_amdgcn_s_wait_asynccnt)
  __builtin_amdgcn_s_wait_asynccnt(0);
#else
  asm volatile("s_wait_asynccnt 0x0" ::: "memory");
#endif
}

__device__ __forceinline__ void async_wait1() {
#if __has_builtin(__builtin_amdgcn_s_wait_asynccnt)
  __builtin_amdgcn_s_wait_asynccnt(1);
#else
  asm volatile("s_wait_asynccnt 0x1" ::: "memory");
#endif
}

// Load a WMMA A/B fragment from a row-major f16 matrix (global or LDS).
// Caller passes p = rowBase + kb + (hi?8:0):
//   elements 0..7  <- p[0..7]    (K = kb + off .. +7)
//   elements 8..15 <- p[16..23]  (K = kb + 16 + off .. +7)
__device__ __forceinline__ v16h frag_load(const _Float16* p) {
  union { v16h v; v8h h[2]; } u;
  u.h[0] = *(const v8h*)(p);
  u.h[1] = *(const v8h*)(p + 16);
  return u.v;
}

__device__ __forceinline__ v8f wmma_f16(v16h a, v16h b, v8f c) {
  return __builtin_amdgcn_wmma_f32_16x16x32_f16(false, a, false, b, (short)0, c,
                                                false, false);
}

__device__ __forceinline__ float gelu_tanh(float x) {
  float x3 = x * x * x;
  return 0.5f * x * (1.0f + tanhf(0.79788456080286536f * (x + 0.044715f * x3)));
}

// ---------------------------------------------------------------------------
// f32 -> f16 copy
// ---------------------------------------------------------------------------
__global__ __launch_bounds__(256) void cvt_f16_kernel(const float* __restrict__ in,
                                                      _Float16* __restrict__ out,
                                                      size_t n) {
  size_t i = (size_t)blockIdx.x * blockDim.x + threadIdx.x;
  if (i < n) out[i] = (_Float16)in[i];
}

// in: K x N row-major f32.  out: N x K row-major f16 (B-operand friendly).
__global__ __launch_bounds__(256) void transpose_f16_kernel(const float* __restrict__ in,
                                                            _Float16* __restrict__ out,
                                                            int K, int N) {
  int i = blockIdx.x * blockDim.x + threadIdx.x;
  if (i < K * N) {
    int k = i / N, n = i % N;
    out[(size_t)n * K + k] = (_Float16)in[i];
  }
}

// ---------------------------------------------------------------------------
// Generic WMMA GEMM:  out[M x N] = A[M x K] * Wt[N x K]^T + bias
// 8 waves/block; wave = 16 rows x 64 cols (4 tiles); K-step 32.
// B tile (64 x 32 f16 = 4 KB) is shared by all waves -> staged in LDS via
// async global->LDS copies, double buffered.
// OUT_MODE: 0 = f32 row-major, 1 = f16 row-major, 2 = f16 per-head transposed
//           Vt[((b*NH+h)*HD+d)*NN + tok]
// ---------------------------------------------------------------------------
template <int ACT_GELU, int OUT_MODE>
__global__ __launch_bounds__(256)
void gemm_wmma_kernel(const _Float16* __restrict__ A, const _Float16* __restrict__ Wt,
                      const float* __restrict__ bias, _Float16* __restrict__ outh,
                      float* __restrict__ outf, int M, int N, int K) {
  __shared__ __align__(16) _Float16 s_b[2][64 * 32];  // 2 x 4 KB

  const int tid  = threadIdx.x;
  const int wave = tid >> 5;
  const int lane = tid & 31;
  const int l16  = lane & 15;
  const int hi   = lane >> 4;
  const int off  = hi ? 8 : 0;
  const int row0 = blockIdx.x * 128 + wave * 16;
  const int col0 = blockIdx.y * 64;
  const int NK   = K >> 5;

  // Each thread stages exactly one 16B chunk of the 64x32 B tile.
  const int rB  = tid >> 2;        // 0..63 : row within tile (column of output)
  const int sB  = (tid & 3) * 8;   // 0,8,16,24 : k-offset in halves
  const _Float16* wrow = Wt + (size_t)(col0 + rB) * K + sB;

  v8f acc[4] = {};
  const _Float16* ap = A + (size_t)(row0 + l16) * K + off;

  stage16(wrow, &s_b[0][rB * 32 + sB]);  // prologue: k-step 0

  for (int i = 0; i < NK; ++i) {
    const int kb = i << 5;
    const int p  = i & 1;
    if (i + 1 < NK) {
      stage16(wrow + ((i + 1) << 5), &s_b[p ^ 1][rB * 32 + sB]);
      async_wait1();
    } else {
      async_wait0();
    }
    __syncthreads();  // buf[p] fully staged by all threads

    v16h a = frag_load(ap + kb);
    if (kb + 64 < K) __builtin_prefetch(ap + kb + 64, 0, 1);  // global_prefetch
#pragma unroll
    for (int t = 0; t < 4; ++t) {
      v16h b = frag_load(&s_b[p][(t * 16 + l16) * 32 + off]);
      acc[t] = wmma_f16(a, b, acc[t]);
    }
    __syncthreads();  // buf[p] consumed; safe to overwrite next iteration
  }

#pragma unroll
  for (int t = 0; t < 4; ++t) {
    const int n  = col0 + t * 16 + l16;
    const float bv = bias[n];
#pragma unroll
    for (int r = 0; r < 8; ++r) {
      const int m = row0 + r + 8 * hi;  // C layout: lanes 0-15 -> M=r, 16-31 -> M=8+r
      float v = acc[t][r] + bv;
      if (ACT_GELU) v = gelu_tanh(v);
      if (OUT_MODE == 0) {
        outf[(size_t)m * N + n] = v;
      } else if (OUT_MODE == 1) {
        outh[(size_t)m * N + n] = (_Float16)v;
      } else {  // per-head transposed V store
        const int bb = m >> 7, tok = m & 127, hh = n >> 5, dd = n & 31;
        outh[(((size_t)bb * NH + hh) * HD + dd) * NN + tok] = (_Float16)v;
      }
    }
  }
}

// ---------------------------------------------------------------------------
// Attention: one block per (b,h). 8 waves; wave w owns query rows 16w..16w+15.
// K tile (128x32) and Vt tile (32x128) are shared by all waves -> staged once
// in LDS via async copies. scores -> softmax -> attn (f32 to d_out, f16 to
// LDS for the C-layout -> A-layout reshape) -> ctx = attn @ V.
// ---------------------------------------------------------------------------
__global__ __launch_bounds__(256)
void attn_kernel(const _Float16* __restrict__ Qh, const _Float16* __restrict__ Kh,
                 const _Float16* __restrict__ Vt, const float* __restrict__ adj,
                 const float* __restrict__ dist, const float* __restrict__ mask,
                 float* __restrict__ attn_out, _Float16* __restrict__ ctx_out) {
  __shared__ __align__(16) _Float16 s_k[NN * HD];     // 8 KB
  __shared__ __align__(16) _Float16 s_v[HD * NN];     // 8 KB
  __shared__ __align__(16) _Float16 s_attn[NN * NN];  // 32 KB

  const int bh = blockIdx.x;
  const int b  = bh >> 3;
  const int h  = bh & 7;
  const int wave = threadIdx.x >> 5;
  const int lane = threadIdx.x & 31;
  const int l16  = lane & 15;
  const int hi   = lane >> 4;
  const int off  = hi ? 8 : 0;
  const int mrow = wave * 16;

  const _Float16* Qb  = Qh + (size_t)b * NN * DM + h * HD;
  const _Float16* Kb  = Kh + (size_t)b * NN * DM + h * HD;
  const _Float16* Vtb = Vt + (size_t)bh * HD * NN;  // contiguous 32x128 tile
  const float* adjb  = adj  + (size_t)b * NN * NN;
  const float* distb = dist + (size_t)b * NN * NN;
  const float* maskb = mask + (size_t)b * NN;

  // ---- stage K (row-major 128x32) and Vt (row-major 32x128) tiles ----
  for (int c = threadIdx.x; c < 512; c += 256) {
    stage16(Kb + (size_t)(c >> 2) * DM + (c & 3) * 8, &s_k[c * 8]);
    stage16(Vtb + c * 8, &s_v[c * 8]);
  }
  async_wait0();
  __syncthreads();

  // ---- scores: K=32 is exactly one WMMA step ----
  v16h qa = frag_load(Qb + (size_t)(mrow + l16) * DM + off);
  v8f s[8];
#pragma unroll
  for (int t = 0; t < 8; ++t) {
    v16h kf = frag_load(&s_k[(t * 16 + l16) * HD + off]);
    v8f z = {};
    s[t] = wmma_f16(qa, kf, z);
  }

  // ---- bias + masked softmax (f32; a row lives across a 16-lane half-group) ----
  const float scale = 0.17677669529663687f;  // 1/sqrt(32)
  float mx[8], sum[8];
#pragma unroll
  for (int r = 0; r < 8; ++r) mx[r] = -3.0e38f;
#pragma unroll
  for (int t = 0; t < 8; ++t) {
    const int n = t * 16 + l16;
    const float mbias = (1.0f - maskb[n]) * -1.0e9f;
#pragma unroll
    for (int r = 0; r < 8; ++r) {
      const int m = mrow + r + 8 * hi;
      float v = s[t][r] * scale + adjb[m * NN + n] + distb[m * NN + n] + mbias;
      s[t][r] = v;
      mx[r] = fmaxf(mx[r], v);
    }
  }
#pragma unroll
  for (int r = 0; r < 8; ++r)
    for (int d = 1; d < 16; d <<= 1) mx[r] = fmaxf(mx[r], __shfl_xor(mx[r], d, 32));
#pragma unroll
  for (int r = 0; r < 8; ++r) sum[r] = 0.0f;
#pragma unroll
  for (int t = 0; t < 8; ++t)
#pragma unroll
    for (int r = 0; r < 8; ++r) {
      float e = __expf(s[t][r] - mx[r]);
      s[t][r] = e;
      sum[r] += e;
    }
#pragma unroll
  for (int r = 0; r < 8; ++r)
    for (int d = 1; d < 16; d <<= 1) sum[r] += __shfl_xor(sum[r], d, 32);

  float* ao = attn_out + (size_t)bh * NN * NN;
#pragma unroll
  for (int r = 0; r < 8; ++r) {
    const float inv = 1.0f / sum[r];
    const int m = mrow + r + 8 * hi;
#pragma unroll
    for (int t = 0; t < 8; ++t) {
      const int n = t * 16 + l16;
      float p = s[t][r] * inv;
      ao[m * NN + n] = p;
      s_attn[m * NN + n] = (_Float16)p;
    }
  }
  __syncthreads();

  // ---- ctx = attn @ V : 16 rows x 32 cols per wave, K=128 in 4 steps ----
  v8f c0 = {}, c1 = {};
  for (int kb = 0; kb < NN; kb += 32) {
    v16h af = frag_load(&s_attn[(mrow + l16) * NN + kb + off]);
    v16h b0 = frag_load(&s_v[l16 * NN + kb + off]);         // col n = l16
    v16h b1 = frag_load(&s_v[(16 + l16) * NN + kb + off]);  // col n = 16+l16
    c0 = wmma_f16(af, b0, c0);
    c1 = wmma_f16(af, b1, c1);
  }

  // ctx stored to (B, N, H*HD) so the O-proj GEMM reads it row-major.
  _Float16* co = ctx_out + (size_t)b * NN * DM + h * HD;
#pragma unroll
  for (int r = 0; r < 8; ++r) {
    const int m = mrow + r + 8 * hi;
    co[(size_t)m * DM + l16]      = (_Float16)c0[r];
    co[(size_t)m * DM + 16 + l16] = (_Float16)c1[r];
  }
}

// ---------------------------------------------------------------------------
// out = Xres + LayerNorm(Y)*g + b  (row width 256, one wave per row)
// ---------------------------------------------------------------------------
__global__ __launch_bounds__(256)
void addln_kernel(const float* __restrict__ Y, const float* __restrict__ Xres,
                  const float* __restrict__ g, const float* __restrict__ bta,
                  float* __restrict__ outf, _Float16* __restrict__ outh) {
  const int wave = threadIdx.x >> 5;
  const int lane = threadIdx.x & 31;
  const size_t row = (size_t)blockIdx.x * 8 + wave;
  const float* y = Y + row * DM;

  float v[8];
  float s = 0.0f;
#pragma unroll
  for (int i = 0; i < 8; ++i) { v[i] = y[lane + 32 * i]; s += v[i]; }
  for (int d = 1; d < 32; d <<= 1) s += __shfl_xor(s, d, 32);
  const float mu = s * (1.0f / DM);

  float var = 0.0f;
#pragma unroll
  for (int i = 0; i < 8; ++i) { float d0 = v[i] - mu; var += d0 * d0; }
  for (int d = 1; d < 32; d <<= 1) var += __shfl_xor(var, d, 32);
  const float rstd = rsqrtf(var * (1.0f / DM) + 1e-5f);

#pragma unroll
  for (int i = 0; i < 8; ++i) {
    const int c = lane + 32 * i;
    float o = Xres[row * DM + c] + (v[i] - mu) * rstd * g[c] + bta[c];
    outf[row * DM + c] = o;
    if (outh) outh[row * DM + c] = (_Float16)o;
  }
}

// ---------------------------------------------------------------------------
extern "C" void kernel_launch(void* const* d_in, const int* in_sizes, int n_in,
                              void* d_out, int out_size, void* d_ws, size_t ws_size,
                              hipStream_t stream) {
  const float* X    = (const float*)d_in[0];
  const float* mask = (const float*)d_in[1];
  const float* adj  = (const float*)d_in[2];
  const float* dist = (const float*)d_in[3];
  const float* Wq = (const float*)d_in[4];  const float* bq  = (const float*)d_in[5];
  const float* Wk = (const float*)d_in[6];  const float* bk  = (const float*)d_in[7];
  const float* Wv = (const float*)d_in[8];  const float* bv  = (const float*)d_in[9];
  const float* Wo = (const float*)d_in[10]; const float* bo  = (const float*)d_in[11];
  const float* g1 = (const float*)d_in[12]; const float* be1 = (const float*)d_in[13];
  const float* W1 = (const float*)d_in[14]; const float* b1  = (const float*)d_in[15];
  const float* W2 = (const float*)d_in[16]; const float* b2  = (const float*)d_in[17];
  const float* g2 = (const float*)d_in[18]; const float* be2 = (const float*)d_in[19];

  const size_t Mtok = (size_t)BZ * NN;  // 32768 tokens
  float* outX    = (float*)d_out;
  float* outAttn = outX + Mtok * DM;    // attn follows X in return order

  // --- workspace carve-out (~237 MB) ---
  char* w = (char*)d_ws;
  auto carve = [&](size_t bytes) -> char* {
    char* p = w;
    w += (bytes + 255) & ~(size_t)255;
    return p;
  };
  _Float16* Xh  = (_Float16*)carve(Mtok * DM * 2);
  _Float16* Qh  = (_Float16*)carve(Mtok * DM * 2);
  _Float16* Kh  = (_Float16*)carve(Mtok * DM * 2);
  _Float16* Vt  = (_Float16*)carve(Mtok * DM * 2);   // (B,H,HD,NN)
  _Float16* Ch  = (_Float16*)carve(Mtok * DM * 2);   // ctx, (B,N,H*HD)
  _Float16* X1h = (_Float16*)carve(Mtok * DM * 2);
  _Float16* Hh  = (_Float16*)carve(Mtok * DF * 2);   // FFN hidden
  _Float16* Wqt = (_Float16*)carve((size_t)DM * DM * 2);
  _Float16* Wkt = (_Float16*)carve((size_t)DM * DM * 2);
  _Float16* Wvt = (_Float16*)carve((size_t)DM * DM * 2);
  _Float16* Wot = (_Float16*)carve((size_t)DM * DM * 2);
  _Float16* W1t = (_Float16*)carve((size_t)DM * DF * 2);
  _Float16* W2t = (_Float16*)carve((size_t)DF * DM * 2);
  float*    Yf  = (float*)carve(Mtok * DM * 4);
  float*    X1f = (float*)carve(Mtok * DM * 4);
  (void)carve(4096);  // pad for prefetch overrun

  // --- stage 0: precision conversion / weight transpose ---
  cvt_f16_kernel<<<(Mtok * DM + 255) / 256, 256, 0, stream>>>(X, Xh, Mtok * DM);
  transpose_f16_kernel<<<(DM * DM + 255) / 256, 256, 0, stream>>>(Wq, Wqt, DM, DM);
  transpose_f16_kernel<<<(DM * DM + 255) / 256, 256, 0, stream>>>(Wk, Wkt, DM, DM);
  transpose_f16_kernel<<<(DM * DM + 255) / 256, 256, 0, stream>>>(Wv, Wvt, DM, DM);
  transpose_f16_kernel<<<(DM * DM + 255) / 256, 256, 0, stream>>>(Wo, Wot, DM, DM);
  transpose_f16_kernel<<<(DM * DF + 255) / 256, 256, 0, stream>>>(W1, W1t, DM, DF);
  transpose_f16_kernel<<<(DF * DM + 255) / 256, 256, 0, stream>>>(W2, W2t, DF, DM);

  // --- stage 1: QKV projections (V stored per-head transposed) ---
  dim3 gD((unsigned)(Mtok / 128), DM / 64);
  gemm_wmma_kernel<0, 1><<<gD, 256, 0, stream>>>(Xh, Wqt, bq, Qh, nullptr,
                                                 (int)Mtok, DM, DM);
  gemm_wmma_kernel<0, 1><<<gD, 256, 0, stream>>>(Xh, Wkt, bk, Kh, nullptr,
                                                 (int)Mtok, DM, DM);
  gemm_wmma_kernel<0, 2><<<gD, 256, 0, stream>>>(Xh, Wvt, bv, Vt, nullptr,
                                                 (int)Mtok, DM, DM);

  // --- stage 2: attention (writes attn to d_out, ctx to ws) ---
  attn_kernel<<<BZ * NH, 256, 0, stream>>>(Qh, Kh, Vt, adj, dist, mask, outAttn, Ch);

  // --- stage 3: output projection + AddNorm ---
  gemm_wmma_kernel<0, 0><<<gD, 256, 0, stream>>>(Ch, Wot, bo, nullptr, Yf,
                                                 (int)Mtok, DM, DM);
  addln_kernel<<<(unsigned)(Mtok / 8), 256, 0, stream>>>(Yf, X, g1, be1, X1f, X1h);

  // --- stage 4: FFN + AddNorm ---
  dim3 gF((unsigned)(Mtok / 128), DF / 64);
  gemm_wmma_kernel<1, 1><<<gF, 256, 0, stream>>>(X1h, W1t, b1, Hh, nullptr,
                                                 (int)Mtok, DF, DM);
  gemm_wmma_kernel<0, 0><<<gD, 256, 0, stream>>>(Hh, W2t, b2, nullptr, Yf,
                                                 (int)Mtok, DM, DF);
  addln_kernel<<<(unsigned)(Mtok / 8), 256, 0, stream>>>(Yf, X1f, g2, be2, outX,
                                                         nullptr);
}
